// FromPredictor_84447646974061
// MI455X (gfx1250) — matrix operand order repaired and checked by reading, not compiled, not run on hardware
//
#include <hip/hip_runtime.h>
#include <hip/hip_bf16.h>

// ---------------------------------------------------------------------------
// MI455X (gfx1250) implementation.
// Compute-bound (~1.16 TFLOP of GEMM vs ~20us of HBM traffic @23.3TB/s):
//  - all large GEMMs via v_wmma_f32_16x16x32_f16 (wave32, f32 accum)
//  - double-buffered LDS tiles filled with global_load_async_to_lds_b128
//    (ASYNCcnt / s_wait_asynccnt), 2-tile-deep pipeline.
// ---------------------------------------------------------------------------

typedef __attribute__((ext_vector_type(16))) _Float16 v16h;
typedef __attribute__((ext_vector_type(8)))  _Float16 v8h;
typedef __attribute__((ext_vector_type(4)))  _Float16 v4h;
typedef __attribute__((ext_vector_type(8)))  float    v8f;

#define BM 128
#define BN 128
#define BK 32

// ---------------------------------------------------------------------------
// Transpose + fp32 -> f16 convert:  W[K][N] -> Wt[Npad][K], zero pad n >= N.
// ---------------------------------------------------------------------------
__global__ __launch_bounds__(256)
void transpose_f16(const float* __restrict__ W, _Float16* __restrict__ Wt,
                   int K, int N, int Npad)
{
    __shared__ float tile[32][33];
    int n0 = blockIdx.x * 32, k0 = blockIdx.y * 32;
    int tx = threadIdx.x & 31, ty = threadIdx.x >> 5;
#pragma unroll
    for (int yy = 0; yy < 4; yy++) {
        int k = k0 + ty + yy * 8, n = n0 + tx;
        tile[ty + yy * 8][tx] = (k < K && n < N) ? W[(size_t)k * N + n] : 0.f;
    }
    __syncthreads();
#pragma unroll
    for (int yy = 0; yy < 4; yy++) {
        int n = n0 + ty + yy * 8, k = k0 + tx;
        if (n < Npad && k < K)
            Wt[(size_t)n * K + k] = (_Float16)tile[tx][ty + yy * 8];
    }
}

// ---------------------------------------------------------------------------
// WMMA GEMM: C[M][ldc] = A[M][K](f16) @ Bt[Npad][K](f16,T) + bias, opt ReLU.
// Requirements: M % 128 == 0, K % 32 == 0, Bt padded to gridDim.x*128 rows.
// Block 256 thr = 8 waves (2 M x 4 N); wave tile 64x32 = 4x2 WMMA tiles.
// Double-buffered LDS, async global->LDS copies (4 b128 per thread per tile).
// ---------------------------------------------------------------------------
template <bool RELU, bool OUTF16>
__global__ __launch_bounds__(256)
void gemm_wmma(const _Float16* __restrict__ A, const _Float16* __restrict__ Bt,
               const float* __restrict__ bias, void* __restrict__ Cout,
               int M, int N, int K, int ldc)
{
    __shared__ _Float16 As[2][BM][BK + 8];   // +16B row pad, 2 x 10KB
    __shared__ _Float16 Bs[2][BM][BK + 8];

    int tid = threadIdx.x;
    int lane = tid & 31, w = tid >> 5;
    int wm = w & 1, wn = w >> 1;
    int blockM = blockIdx.y * BM, blockN = blockIdx.x * BN;

    int row = tid >> 2, ch = tid & 3;        // staging: (row, 16B chunk)

    // Issue one 128x32 A tile + B tile into LDS buffer `buf` (async).
    auto issue = [&](int buf, int k0) {
        const _Float16* gA0 = A  + (size_t)(blockM + row) * K + k0 + ch * 8;
        const _Float16* gA1 = gA0 + (size_t)64 * K;
        const _Float16* gB0 = Bt + (size_t)(blockN + row) * K + k0 + ch * 8;
        const _Float16* gB1 = gB0 + (size_t)64 * K;
        unsigned lA0 = (unsigned)(size_t)&As[buf][row][ch * 8];
        unsigned lA1 = (unsigned)(size_t)&As[buf][row + 64][ch * 8];
        unsigned lB0 = (unsigned)(size_t)&Bs[buf][row][ch * 8];
        unsigned lB1 = (unsigned)(size_t)&Bs[buf][row + 64][ch * 8];
        asm volatile("global_load_async_to_lds_b128 %0, %1, off"
                     :: "v"(lA0), "v"((unsigned long long)(size_t)gA0) : "memory");
        asm volatile("global_load_async_to_lds_b128 %0, %1, off"
                     :: "v"(lA1), "v"((unsigned long long)(size_t)gA1) : "memory");
        asm volatile("global_load_async_to_lds_b128 %0, %1, off"
                     :: "v"(lB0), "v"((unsigned long long)(size_t)gB0) : "memory");
        asm volatile("global_load_async_to_lds_b128 %0, %1, off"
                     :: "v"(lB1), "v"((unsigned long long)(size_t)gB1) : "memory");
    };

    int nt = K / BK;
    issue(0, 0);
    if (nt > 1) issue(1, BK);

    v8f acc[4][2] = {};

    int rr = lane & 15;
    int kb  = (lane < 16) ? 0 : 8;    // A frag: interleaved halves (ISA 7.12.2)
    int kgB = (lane < 16) ? 0 : 8;    // B frag: interleaved halves, same as A

    for (int n = 0; n < nt; n++) {
        // Drain the oldest tile: 4 async ops/thread/tile, in-order completion,
        // so <=4 outstanding means tile n has landed in LDS.
        if (n + 1 < nt) asm volatile("s_wait_asynccnt 4" ::: "memory");
        else            asm volatile("s_wait_asynccnt 0" ::: "memory");
        __syncthreads();

        int buf = n & 1;
        union U { v16h v; v8h h[2]; };
        U af[4], bf[2];
#pragma unroll
        for (int mt = 0; mt < 4; mt++) {
            int rowA = wm * 64 + mt * 16 + rr;
            af[mt].h[0] = *(const v8h*)&As[buf][rowA][kb];
            af[mt].h[1] = *(const v8h*)&As[buf][rowA][kb + 16];
        }
#pragma unroll
        for (int nt2 = 0; nt2 < 2; nt2++) {
            int rowB = wn * 32 + nt2 * 16 + rr;
            bf[nt2].h[0] = *(const v8h*)&Bs[buf][rowB][kgB];
            bf[nt2].h[1] = *(const v8h*)&Bs[buf][rowB][kgB + 16];
        }
#pragma unroll
        for (int mt = 0; mt < 4; mt++)
#pragma unroll
            for (int nt2 = 0; nt2 < 2; nt2++)
                acc[mt][nt2] = __builtin_amdgcn_wmma_f32_16x16x32_f16(
                    false, af[mt].v, false, bf[nt2].v, (short)0, acc[mt][nt2],
                    false, false);

        __syncthreads();          // all reads of buf done before refill
        if (n + 2 < nt) issue(buf, (n + 2) * BK);
    }

    float* Cf = (float*)Cout;
    _Float16* Ch = (_Float16*)Cout;
    int rhi = (lane & 16) ? 8 : 0;
    int cc = lane & 15;
#pragma unroll
    for (int mt = 0; mt < 4; mt++)
#pragma unroll
        for (int nt2 = 0; nt2 < 2; nt2++) {
            int gcol = blockN + wn * 32 + nt2 * 16 + cc;
            if (gcol >= N) continue;
            float bv = bias ? bias[gcol] : 0.f;
#pragma unroll
            for (int r2 = 0; r2 < 8; r2++) {
                int grow = blockM + wm * 64 + mt * 16 + rhi + r2;
                if (grow >= M) continue;
                float v = acc[mt][nt2][r2] + bv;
                if (RELU) v = fmaxf(v, 0.f);
                size_t o = (size_t)grow * ldc + gcol;
                if (OUTF16) Ch[o] = (_Float16)v; else Cf[o] = v;
            }
        }
}

// ---------------------------------------------------------------------------
// Per (span, head) attention: L=16 tokens, DK=DV=128. One wave per block.
// Scores via 4x WMMA (K=128), in-register softmax, VALU P@V.
// q/k/v/o layout: [token32768][H*128] f16.
// ---------------------------------------------------------------------------
__global__ __launch_bounds__(32)
void attn16(const _Float16* __restrict__ qh, const _Float16* __restrict__ kh,
            const _Float16* __restrict__ vh, _Float16* __restrict__ oh)
{
    int hd = blockIdx.x;     // head 0..2
    int span = blockIdx.y;   // 0..2047
    int lane = threadIdx.x;
    int r = lane & 15;
    int kb  = (lane < 16) ? 0 : 8;
    int kgB = (lane < 16) ? 0 : 8;
    size_t base = ((size_t)span * 16) * 384 + hd * 128;

    v8f acc = {};
#pragma unroll
    for (int kk = 0; kk < 4; kk++) {
        union U { v16h v; v8h h2[2]; } a, b;
        size_t ao = base + (size_t)r * 384 + kk * 32;
        a.h2[0] = *(const v8h*)(qh + ao + kb);
        a.h2[1] = *(const v8h*)(qh + ao + kb + 16);
        b.h2[0] = *(const v8h*)(kh + ao + kgB);
        b.h2[1] = *(const v8h*)(kh + ao + kgB + 16);
        acc = __builtin_amdgcn_wmma_f32_16x16x32_f16(false, a.v, false, b.v,
                                                     (short)0, acc, false, false);
    }

    __shared__ float Ps[16][17];
    const float sc = 0.08838834764831845f;   // 1/sqrt(128)
    int rowhi = (lane < 16) ? 0 : 8;
#pragma unroll
    for (int i = 0; i < 8; i++) {
        float v = acc[i] * sc;
        float m = v;
        for (int off = 8; off; off >>= 1) m = fmaxf(m, __shfl_xor(m, off, 32));
        float e = __expf(v - m);
        float s = e;
        for (int off = 8; off; off >>= 1) s += __shfl_xor(s, off, 32);
        Ps[rowhi + i][r] = e / s;
    }
    __syncthreads();

    int d0 = lane * 4;   // 128 dims / 32 lanes
    for (int t = 0; t < 16; t++) {
        float o0 = 0, o1 = 0, o2 = 0, o3 = 0;
#pragma unroll
        for (int k = 0; k < 16; k++) {
            float p = Ps[t][k];
            v4h vk = *(const v4h*)(vh + base + (size_t)k * 384 + d0);
            o0 += p * (float)vk[0]; o1 += p * (float)vk[1];
            o2 += p * (float)vk[2]; o3 += p * (float)vk[3];
        }
        v4h ov = { (_Float16)o0, (_Float16)o1, (_Float16)o2, (_Float16)o3 };
        *(v4h*)(oh + base + (size_t)t * 384 + d0) = ov;
    }
}

// ---------------------------------------------------------------------------
// x = LN(x + y) ; also writes f16 mirror. One block per token row (D=768).
// ---------------------------------------------------------------------------
__global__ __launch_bounds__(256)
void add_ln(const float* __restrict__ yin, float* __restrict__ x,
            _Float16* __restrict__ xh, const float* __restrict__ g,
            const float* __restrict__ bb)
{
    size_t row = blockIdx.x;
    int tid = threadIdx.x, lane = tid & 31, w = tid >> 5;
    const float* yr = yin + row * 768;
    float* xr = x + row * 768;
    float r[3]; float s = 0.f, sq = 0.f;
#pragma unroll
    for (int j = 0; j < 3; j++) {
        int i = tid + j * 256;
        r[j] = xr[i] + yr[i];
        s += r[j]; sq += r[j] * r[j];
    }
    for (int off = 16; off; off >>= 1) {
        s  += __shfl_xor(s, off, 32);
        sq += __shfl_xor(sq, off, 32);
    }
    __shared__ float ls[8], lq[8];
    if (lane == 0) { ls[w] = s; lq[w] = sq; }
    __syncthreads();
    if (tid == 0) {
        float ts = 0, tq = 0;
        for (int i = 0; i < 8; i++) { ts += ls[i]; tq += lq[i]; }
        ls[0] = ts; lq[0] = tq;
    }
    __syncthreads();
    float mean = ls[0] * (1.f / 768.f);
    float var  = lq[0] * (1.f / 768.f) - mean * mean;
    float inv  = rsqrtf(var + 1e-5f);
#pragma unroll
    for (int j = 0; j < 3; j++) {
        int i = tid + j * 256;
        float o = (r[j] - mean) * inv * g[i] + bb[i];
        xr[i] = o;
        xh[row * 768 + i] = (_Float16)o;
    }
}

// ---------------------------------------------------------------------------
// Gather entity spans -> x[2048*16][768] (f32 + f16), masked rows zeroed.
// ---------------------------------------------------------------------------
__global__ __launch_bounds__(256)
void gather_spans(const float* __restrict__ q_enc, const int* __restrict__ ranges,
                  float* __restrict__ x, _Float16* __restrict__ xh)
{
    int span = blockIdx.x;        // b*32 + e
    int b = span >> 5;
    int start = ranges[span * 2 + 0];
    int len = ranges[span * 2 + 1] - start;
    for (int i = threadIdx.x; i < 16 * 768; i += 256) {
        int l = i / 768, d = i - l * 768;
        int src = start + l; src = src < 0 ? 0 : (src > 511 ? 511 : src);
        float v = (l < len) ? q_enc[((size_t)b * 512 + src) * 768 + d] : 0.f;
        size_t o = (size_t)span * 16 * 768 + i;
        x[o] = v; xh[o] = (_Float16)v;
    }
}

__global__ __launch_bounds__(256)
void gather_ent(const float* __restrict__ x, float* __restrict__ ent,
                _Float16* __restrict__ enth)
{
    size_t i = (size_t)blockIdx.x * 256 + threadIdx.x;   // 2048*768
    if (i >= (size_t)2048 * 768) return;
    size_t span = i / 768, d = i - span * 768;
    float v = x[(span * 16) * 768 + d];
    ent[i] = v; enth[i] = (_Float16)v;
}

// ---------------------------------------------------------------------------
// Cross attention per (b,t): scores over E=32 entities (dim 770), softmax,
// weighted sum of enc_ent, masked by t < q_q_len[b].
// ---------------------------------------------------------------------------
__global__ __launch_bounds__(256)
void cross_attn(const float* __restrict__ q_enc, const float* __restrict__ hint,
                const int* __restrict__ qlen, const float* __restrict__ entp,
                const float* __restrict__ ent, float* __restrict__ wtd)
{
    int t = blockIdx.x, b = blockIdx.y;
    int tid = threadIdx.x, lane = tid & 31, w = tid >> 5;
    __shared__ float s_sc[32], s_p[32];
    const float* qrow = q_enc + ((size_t)b * 512 + t) * 768;
    float h0 = hint[((size_t)b * 128 + t) * 2 + 0];
    float h1 = hint[((size_t)b * 128 + t) * 2 + 1];
    for (int e = w; e < 32; e += 8) {
        const float* prow = entp + ((size_t)b * 32 + e) * 770;
        float part = 0.f;
        for (int i = lane; i < 768; i += 32) part += qrow[i] * prow[i];
        if (lane == 0) part += h0 * prow[768] + h1 * prow[769];
        for (int off = 16; off; off >>= 1) part += __shfl_xor(part, off, 32);
        if (lane == 0) s_sc[e] = part;
    }
    __syncthreads();
    if (tid < 32) {
        float v = s_sc[tid];
        float m = v;
        for (int off = 16; off; off >>= 1) m = fmaxf(m, __shfl_xor(m, off, 32));
        float e = __expf(v - m);
        float s = e;
        for (int off = 16; off; off >>= 1) s += __shfl_xor(s, off, 32);
        s_p[tid] = e / s;
    }
    __syncthreads();
    float scale = (t < qlen[b]) ? 1.f : 0.f;
    for (int d = tid; d < 768; d += 256) {
        float a = 0.f;
#pragma unroll 8
        for (int e = 0; e < 32; e++)
            a += s_p[e] * ent[((size_t)b * 32 + e) * 768 + d];
        wtd[((size_t)(b * 128 + t)) * 768 + d] = a * scale;
    }
}

__global__ __launch_bounds__(256)
void reduce_wa(const float* __restrict__ wtd, float* __restrict__ wa)
{
    int d = blockIdx.x * 256 + threadIdx.x;   // 0..767 (grid.x = 3)
    int b = blockIdx.y;
    float s = 0.f;
    for (int t = 0; t < 128; t++) s += wtd[((size_t)(b * 128 + t)) * 768 + d];
    wa[(size_t)b * 768 + d] = s;
}

// ---------------------------------------------------------------------------
// feat = [q_enc[b,0,:], wa[b,:]] (1536) -> relu(@W1+b1) (1024) -> @W2+b2.
// ---------------------------------------------------------------------------
__global__ __launch_bounds__(256)
void final_mlp(const float* __restrict__ q_enc, const float* __restrict__ wa,
               const float* __restrict__ W1, const float* __restrict__ b1,
               const float* __restrict__ W2, const float* __restrict__ b2,
               float* __restrict__ out)
{
    int b = blockIdx.x, tid = threadIdx.x;
    __shared__ float feat[1536];
    __shared__ float red[256];
    for (int i = tid; i < 768; i += 256) {
        feat[i]       = q_enc[((size_t)b * 512) * 768 + i];
        feat[768 + i] = wa[(size_t)b * 768 + i];
    }
    __syncthreads();
    float part = 0.f;
#pragma unroll
    for (int jj = 0; jj < 4; jj++) {
        int j = tid + jj * 256;
        float acc = b1[j];
        for (int i = 0; i < 1536; i++) acc += feat[i] * W1[(size_t)i * 1024 + j];
        acc = fmaxf(acc, 0.f);
        part += acc * W2[j];
    }
    red[tid] = part;
    __syncthreads();
    for (int s = 128; s; s >>= 1) {
        if (tid < s) red[tid] += red[tid + s];
        __syncthreads();
    }
    if (tid == 0) out[b] = red[0] + b2[0];
}

// ---------------------------------------------------------------------------
extern "C" void kernel_launch(void* const* d_in, const int* in_sizes, int n_in,
                              void* d_out, int out_size, void* d_ws, size_t ws_size,
                              hipStream_t stream)
{
    (void)in_sizes; (void)n_in; (void)out_size; (void)ws_size;

    const float* q_enc  = (const float*)d_in[0];
    const int*   q_qlen = (const int*)  d_in[1];
    const float* hint   = (const float*)d_in[2];
    const int*   ranges = (const int*)  d_in[3];
    const float* Wq  = (const float*)d_in[4];   const float* bq  = (const float*)d_in[5];
    const float* Wk  = (const float*)d_in[6];   const float* bk  = (const float*)d_in[7];
    const float* Wv  = (const float*)d_in[8];   const float* bv  = (const float*)d_in[9];
    const float* Wo  = (const float*)d_in[10];  const float* bo  = (const float*)d_in[11];
    const float* l1g = (const float*)d_in[12];  const float* l1b = (const float*)d_in[13];
    const float* Wf1 = (const float*)d_in[14];  const float* bf1 = (const float*)d_in[15];
    const float* Wf2 = (const float*)d_in[16];  const float* bf2 = (const float*)d_in[17];
    const float* l2g = (const float*)d_in[18];  const float* l2b = (const float*)d_in[19];
    const float* Wea = (const float*)d_in[20];  const float* bea = (const float*)d_in[21];
    const float* W1  = (const float*)d_in[22];  const float* b1  = (const float*)d_in[23];
    const float* W2  = (const float*)d_in[24];  const float* b2  = (const float*)d_in[25];

    // ---- workspace layout ----
    constexpr size_t TOK = 32768;   // 2048 spans * 16 tokens
    char* p = (char*)d_ws;
    size_t off = 0;
    auto alloc = [&](size_t bytes) -> void* {
        void* q = p + off; off += (bytes + 255) & ~(size_t)255; return q;
    };
    float*     x    = (float*)    alloc(TOK * 768 * 4);
    float*     yb   = (float*)    alloc(TOK * 768 * 4);
    _Float16*  xh   = (_Float16*) alloc(TOK * 768 * 2);
    char*      big  = (char*)     alloc(TOK * 3072 * 2);  // ffn hidden / qkv+o alias
    _Float16*  qh = (_Float16*)big;
    _Float16*  kh = qh + TOK * 384;
    _Float16*  vh = kh + TOK * 384;
    _Float16*  oh = vh + TOK * 384;
    _Float16*  h1 = (_Float16*)big;                        // reuses qkv/o region
    float*     ent  = (float*)    alloc((size_t)2048 * 768 * 4);
    _Float16*  enth = (_Float16*) alloc((size_t)2048 * 768 * 2);
    float*     entp = (float*)    alloc((size_t)2048 * 770 * 4);
    float*     wtd  = (float*)    alloc((size_t)64 * 128 * 768 * 4);
    float*     wa   = (float*)    alloc((size_t)64 * 768 * 4);
    _Float16 *WqT[3], *WkT[3], *WvT[3], *WoT[3], *Wf1T[3], *Wf2T[3];
    for (int i = 0; i < 3; i++) {
        WqT[i]  = (_Float16*)alloc((size_t)384 * 768 * 2);
        WkT[i]  = (_Float16*)alloc((size_t)384 * 768 * 2);
        WvT[i]  = (_Float16*)alloc((size_t)384 * 768 * 2);
        WoT[i]  = (_Float16*)alloc((size_t)768 * 384 * 2);
        Wf1T[i] = (_Float16*)alloc((size_t)3072 * 768 * 2);
        Wf2T[i] = (_Float16*)alloc((size_t)768 * 3072 * 2);
    }
    _Float16* WeaT = (_Float16*)alloc((size_t)896 * 768 * 2);

    // ---- weight transpose/convert (fp32 -> f16, B-transposed, padded) ----
    for (int i = 0; i < 3; i++) {
        transpose_f16<<<dim3(12, 24), 256, 0, stream>>>(Wq  + (size_t)i * 768 * 384, WqT[i],  768, 384, 384);
        transpose_f16<<<dim3(12, 24), 256, 0, stream>>>(Wk  + (size_t)i * 768 * 384, WkT[i],  768, 384, 384);
        transpose_f16<<<dim3(12, 24), 256, 0, stream>>>(Wv  + (size_t)i * 768 * 384, WvT[i],  768, 384, 384);
        transpose_f16<<<dim3(24, 12), 256, 0, stream>>>(Wo  + (size_t)i * 384 * 768, WoT[i],  384, 768, 768);
        transpose_f16<<<dim3(96, 24), 256, 0, stream>>>(Wf1 + (size_t)i * 768 * 3072, Wf1T[i], 768, 3072, 3072);
        transpose_f16<<<dim3(24, 96), 256, 0, stream>>>(Wf2 + (size_t)i * 3072 * 768, Wf2T[i], 3072, 768, 768);
    }
    transpose_f16<<<dim3(28, 24), 256, 0, stream>>>(Wea, WeaT, 768, 770, 896);

    // ---- span gather ----
    gather_spans<<<2048, 256, 0, stream>>>(q_enc, ranges, x, xh);

    // ---- 3 transformer layers ----
    for (int i = 0; i < 3; i++) {
        gemm_wmma<false, true><<<dim3(3, 256), 256, 0, stream>>>(xh, WqT[i], bq + (size_t)i * 384, qh, 32768, 384, 768, 384);
        gemm_wmma<false, true><<<dim3(3, 256), 256, 0, stream>>>(xh, WkT[i], bk + (size_t)i * 384, kh, 32768, 384, 768, 384);
        gemm_wmma<false, true><<<dim3(3, 256), 256, 0, stream>>>(xh, WvT[i], bv + (size_t)i * 384, vh, 32768, 384, 768, 384);
        attn16<<<dim3(3, 2048), 32, 0, stream>>>(qh, kh, vh, oh);
        gemm_wmma<false, false><<<dim3(6, 256), 256, 0, stream>>>(oh, WoT[i], bo + (size_t)i * 768, yb, 32768, 768, 384, 768);
        add_ln<<<32768, 256, 0, stream>>>(yb, x, xh, l1g + (size_t)i * 768, l1b + (size_t)i * 768);
        gemm_wmma<true, true><<<dim3(24, 256), 256, 0, stream>>>(xh, Wf1T[i], bf1 + (size_t)i * 3072, h1, 32768, 3072, 768, 3072);
        gemm_wmma<false, false><<<dim3(6, 256), 256, 0, stream>>>(h1, Wf2T[i], bf2 + (size_t)i * 768, yb, 32768, 768, 3072, 768);
        add_ln<<<32768, 256, 0, stream>>>(yb, x, xh, l2g + (size_t)i * 768, l2b + (size_t)i * 768);
    }

    // ---- entity heads + cross attention tail ----
    gather_ent<<<6144, 256, 0, stream>>>(x, ent, enth);
    gemm_wmma<false, false><<<dim3(7, 16), 256, 0, stream>>>(enth, WeaT, bea, entp, 2048, 770, 768, 770);
    cross_attn<<<dim3(128, 64), 256, 0, stream>>>(q_enc, hint, q_qlen, entp, ent, wtd);
    reduce_wa<<<dim3(3, 64), 256, 0, stream>>>(wtd, wa);
    final_mlp<<<64, 256, 0, stream>>>(q_enc, wa, W1, b1, W2, b2, (float*)d_out);
}
